// Clustering_56521769615815
// MI455X (gfx1250) — matrix operand order repaired and checked by reading, not compiled, not run on hardware
//
#include <hip/hip_runtime.h>
#include <hip/hip_bf16.h>
#include <math.h>

typedef __attribute__((ext_vector_type(16))) _Float16 v16h;
typedef __attribute__((ext_vector_type(8)))  _Float16 v8h;
typedef __attribute__((ext_vector_type(8)))  float    v8f;

// ---------------------------------------------------------------------------
// Kernel 1/3: fused 1x1 conv(s) as WMMA GEMM.
//   out[cv][o][col] = sum_i w[cv][o][i] * x[i][col] + b[cv][o]
//   cols = (batch, pixel) pairs; channel stride in memory = 16384 floats.
//   M=96 per conv, K=96 (3 wmma k-steps of 32), N tiled 16 per wave.
// Block = 256 threads = 8 waves, each wave does 16 columns -> 128 cols/block.
// All conv-index selection is compile-time-foldable so every load/store
// traces to a kernarg pointer (global address space, not flat).
// ---------------------------------------------------------------------------
template <int NCONV>
__global__ __launch_bounds__(256) void conv1x1_wmma(
    const float* __restrict__ x,
    const float* __restrict__ w0, const float* __restrict__ bb0,
    const float* __restrict__ w1, const float* __restrict__ bb1,
    const float* __restrict__ w2, const float* __restrict__ bb2,
    float* __restrict__ o0, float* __restrict__ o1, float* __restrict__ o2)
{
    __shared__ __align__(16) _Float16 wlds[NCONV * 96 * 96];
    __shared__ float blds[NCONV * 96];

    const int tid  = threadIdx.x;
    const int wid  = tid >> 5;
    const int lane = tid & 31;

    // Wave's 16-column group.
    const int col0 = blockIdx.x * 128 + wid * 16;
    const int bimg = col0 >> 14;          // 16384 pixels per image
    const int p0   = col0 & 16383;
    const int px   = p0 + (lane & 15);
    const float* xcol = x + (size_t)bimg * 96 * 16384 + px;

    // Warm the pixel columns while we stage weights.
    __builtin_prefetch(xcol, 0, 0);
    __builtin_prefetch(xcol + (size_t)48 * 16384, 0, 0);

    // Stage weights (f32 -> f16) and biases into LDS (direct kernarg refs).
    for (int idx = tid; idx < 9216; idx += 256) {
        wlds[idx] = (_Float16)w0[idx];
        if (NCONV > 1) {
            wlds[9216 + idx]     = (_Float16)w1[idx];
            wlds[2 * 9216 + idx] = (_Float16)w2[idx];
        }
    }
    if (tid < 96) {
        blds[tid] = bb0[tid];
        if (NCONV > 1) { blds[96 + tid] = bb1[tid]; blds[192 + tid] = bb2[tid]; }
    }
    __syncthreads();

    // ---- B fragments: contiguous k order, k = 16*h + e (same as A) ----
    // lanes 0-15 : K = k0..k0+15  of column N=lane
    // lanes 16-31: K = k0+16..k0+31 of column N=lane-16
    const int khalf = (lane & 16) ? 16 : 0;
    v16h bfr[3];
#pragma unroll
    for (int ks = 0; ks < 3; ++ks) {
        const int kb = ks * 32 + khalf;
        v16h bf;
#pragma unroll
        for (int e = 0; e < 16; ++e)
            bf[e] = (_Float16)xcol[(size_t)(kb + e) * 16384];
        bfr[ks] = bf;
    }

    // ---- A fragments: contiguous k order, k = 16*h + e (same as B) ----
    // lanes 0-15 : elems 0..15 = K 0..15  (row M=lane)
    // lanes 16-31: elems 0..15 = K 16..31 (row M=lane-16)
    const int rowA = lane & 15;
    const int aofs = (lane & 16) ? 16 : 0;
    const int oofs = (lane & 16) ? 8 : 0;   // C/D: vgpr r -> row r (lo) / r+8 (hi)

#pragma unroll
    for (int cv = 0; cv < NCONV; ++cv) {
        // Constant-foldable selection after unroll -> global addr space.
        const _Float16* wl = &wlds[cv * 9216];
        float* outp = (cv == 0) ? o0 : ((cv == 1) ? o1 : o2);
        float* opBase = outp + ((size_t)bimg * 96 + oofs) * 16384 + px;
#pragma unroll
        for (int mt = 0; mt < 6; ++mt) {
            const int m0 = mt * 16;
            v8f acc;
#pragma unroll
            for (int r = 0; r < 8; ++r)
                acc[r] = blds[cv * 96 + m0 + oofs + r];
#pragma unroll
            for (int ks = 0; ks < 3; ++ks) {
                const int kA = ks * 32 + aofs;
                const v8h lo = *(const v8h*)(wl + (m0 + rowA) * 96 + kA);
                const v8h hi = *(const v8h*)(wl + (m0 + rowA) * 96 + kA + 8);
                v16h a;
#pragma unroll
                for (int e = 0; e < 8; ++e) { a[e] = lo[e]; a[e + 8] = hi[e]; }
                acc = __builtin_amdgcn_wmma_f32_16x16x32_f16(
                    false, a, false, bfr[ks], (short)0, acc, false, false);
            }
            float* op = opBase + (size_t)m0 * 16384;
#pragma unroll
            for (int r = 0; r < 8; ++r)
                op[(size_t)r * 16384] = acc[r];
        }
    }
}

// ---------------------------------------------------------------------------
// Kernel 2: clustering. One workgroup (256 threads) per b' slice (256 slices).
// Each slice is a contiguous 98304-float chunk of xp / value / feature.
// Views inside a slice:
//   channel-major: t[c*4096 + n]   (c<24, n<4096, n = w*64+h)
//   row view     : t[n*24 + cc]    (raw reshape [4096,24])
// Result overwrites the `value` slice in place (value fully consumed first).
// ---------------------------------------------------------------------------
__global__ __launch_bounds__(256) void cluster_kernel(
    const float* __restrict__ xp, float* __restrict__ val,
    const float* __restrict__ feat,
    const float* __restrict__ wc, const float* __restrict__ bc,
    const float* __restrict__ alphap, const float* __restrict__ betap)
{
    __shared__ float poolx[96], poolf[96], cent[96], cnew[96], cnv[96], omc[96];
    __shared__ float cnt[4];
    __shared__ float simv[4096];
    __shared__ int   simi[4096];

    const int b = blockIdx.x, tid = threadIdx.x;
    const float* xpB = xp  + (size_t)b * 98304;
    const float* vB  = val + (size_t)b * 98304;
    const float* fB  = feat + (size_t)b * 98304;
    float*       oB  = val + (size_t)b * 98304;   // in-place output

    // (a) adaptive_pool(xp) and adaptive_pool(feature) to [24,2,2]
    if (tid < 192) {
        const int g = tid % 96;
        const float* src = (tid < 96) ? xpB : fB;
        const int c = g >> 2, q = g & 3, i = q >> 1, j = q & 1;
        const float* base = src + c * 4096 + (i * 32) * 64 + j * 32;
        float s = 0.f;
        for (int wi = 0; wi < 32; ++wi) {
            const float* row = base + wi * 64;
            for (int hj = 0; hj < 32; ++hj) s += row[hj];
        }
        s *= (1.0f / 1024.0f);
        if (tid < 96) poolx[g] = s; else poolf[g] = s;
    }
    if (tid < 4)  cnt[tid] = 0.f;
    if (tid < 96) cnew[tid] = 0.f;
    __syncthreads();

    // (b) centers = conv1x1(poolx, wc, bc); flat [o*4+q] == centers[m*24+cc]
    if (tid < 96) {
        const int o = tid >> 2, q = tid & 3;
        float s = bc[o];
        for (int i = 0; i < 24; ++i) s += wc[o * 24 + i] * poolx[i * 4 + q];
        cent[tid] = s;
    }
    __syncthreads();

    // (c) sim1 = softmax_m(centers . value[n]); cnew[m][cc] += sim1*feature
    for (int k = 0; k < 16; ++k) {
        const int n = tid + k * 256;
        float vr[24];
#pragma unroll
        for (int cc = 0; cc < 24; ++cc) vr[cc] = vB[n * 24 + cc];
        float d[4];
#pragma unroll
        for (int m = 0; m < 4; ++m) {
            float s = 0.f;
#pragma unroll
            for (int cc = 0; cc < 24; ++cc) s += cent[m * 24 + cc] * vr[cc];
            d[m] = s;
        }
        const float mx = fmaxf(fmaxf(d[0], d[1]), fmaxf(d[2], d[3]));
        float sm[4]; float ssum = 0.f;
#pragma unroll
        for (int m = 0; m < 4; ++m) { sm[m] = __expf(d[m] - mx); ssum += sm[m]; }
        const float inv = 1.0f / ssum;
        float fr[24];
#pragma unroll
        for (int cc = 0; cc < 24; ++cc) fr[cc] = fB[n * 24 + cc];
#pragma unroll
        for (int m = 0; m < 4; ++m) {
            const float w = sm[m] * inv;
#pragma unroll
            for (int cc = 0; cc < 24; ++cc)
                atomicAdd(&cnew[m * 24 + cc], w * fr[cc]);
        }
    }
    __syncthreads();

    // (d) cn[m][cc] = cnew_flat[cc*4+m] / ||.||   (transpose of [c,M] view)
    if (tid < 4) {
        const int m = tid;
        float s = 0.f;
        for (int cc = 0; cc < 24; ++cc) { float v = cnew[cc * 4 + m]; s += v * v; }
        const float nrm = fmaxf(sqrtf(s), 1e-12f);
        for (int cc = 0; cc < 24; ++cc) cnv[m * 24 + cc] = cnew[cc * 4 + m] / nrm;
    }
    if (tid < 96) omc[tid] = 0.f;
    __syncthreads();

    const float sa = alphap[0], sb = betap[0];

    // (e) sim2 = sigmoid(beta + alpha * cn . xn); hard argmax over m
    for (int k = 0; k < 16; ++k) {
        const int n = tid + k * 256;
        float xv[24]; float s2 = 0.f;
#pragma unroll
        for (int cc = 0; cc < 24; ++cc) {
            xv[cc] = xpB[cc * 4096 + n];
            s2 += xv[cc] * xv[cc];
        }
        const float invn = 1.0f / fmaxf(sqrtf(s2), 1e-12f);
        float best = -1.f; int bm = 0;
#pragma unroll
        for (int m = 0; m < 4; ++m) {
            float dt = 0.f;
#pragma unroll
            for (int cc = 0; cc < 24; ++cc) dt += cnv[m * 24 + cc] * xv[cc];
            const float t = sb + sa * dt * invn;
            const float sg = 1.0f / (1.0f + __expf(-t));
            if (sg > best) { best = sg; bm = m; }   // ties -> first (argmax)
        }
        simv[n] = best;
        simi[n] = bm;
        atomicAdd(&cnt[bm], 1.0f);
    }
    __syncthreads();

    // (f) omc[m][cc] = (sum_n sim2*feature + centers_feature) / (cnt+1)
    for (int k = 0; k < 16; ++k) {
        const int n = tid + k * 256;
        const int m = simi[n];
        const float v = simv[n];
        float fr[24];
#pragma unroll
        for (int cc = 0; cc < 24; ++cc) fr[cc] = fB[n * 24 + cc];
#pragma unroll
        for (int cc = 0; cc < 24; ++cc)
            atomicAdd(&omc[m * 24 + cc], v * fr[cc]);
    }
    __syncthreads();
    if (tid < 96)
        omc[tid] = (omc[tid] + poolf[tid]) / (cnt[tid / 24] + 1.0f);
    __syncthreads();

    // (g) out[n][cc] = omc[idx[n]][cc] * sim2[idx[n]][n]  (one-hot collapse)
    for (int k = 0; k < 16; ++k) {
        const int n = tid + k * 256;
        const int m = simi[n];
        const float v = simv[n];
#pragma unroll
        for (int cc = 0; cc < 24; ++cc)
            oB[n * 24 + cc] = omc[m * 24 + cc] * v;
    }
}

// ---------------------------------------------------------------------------
extern "C" void kernel_launch(void* const* d_in, const int* in_sizes, int n_in,
                              void* d_out, int out_size, void* d_ws, size_t ws_size,
                              hipStream_t stream) {
    const float* x  = (const float*)d_in[0];
    const float* w1 = (const float*)d_in[1];
    const float* b1 = (const float*)d_in[2];
    const float* wv = (const float*)d_in[3];
    const float* bv = (const float*)d_in[4];
    const float* wf = (const float*)d_in[5];
    const float* bf = (const float*)d_in[6];
    const float* wc = (const float*)d_in[7];
    const float* bc = (const float*)d_in[8];
    const float* w2 = (const float*)d_in[9];
    const float* b2 = (const float*)d_in[10];
    const float* sa = (const float*)d_in[11];
    const float* sb = (const float*)d_in[12];
    float* out = (float*)d_out;

    const size_t T = 25165824;            // 16*96*128*128 floats
    float* xp   = (float*)d_ws;           // [16,96,128,128]
    float* val  = xp + T;                 // value; later reused as cluster output
    float* feat = val + T;

    // Stage 1: fused xp/value/feature convs (WMMA GEMM), 262144 cols / 128.
    conv1x1_wmma<3><<<2048, 256, 0, stream>>>(x, w1, b1, wv, bv, wf, bf,
                                              xp, val, feat);
    // Stage 2: per-slice clustering; writes result over `val` in place.
    cluster_kernel<<<256, 256, 0, stream>>>(xp, val, feat, wc, bc, sa, sb);
    // Stage 3: final conv with w2 (WMMA GEMM) -> d_out.
    conv1x1_wmma<1><<<2048, 256, 0, stream>>>(val, w2, b2, w2, b2, w2, b2,
                                              out, out, out);
}
